// HeteroGNN_5643587027022
// MI455X (gfx1250) — compile-verified, matrix operand order checked by reading, not run on hardware
//
#include <hip/hip_runtime.h>

// ---------------------------------------------------------------------------
// HeteroGNN on MI455X (gfx1250, wave32, WMMA).
// Activations f16, WMMA f32_16x16x32_f16 for all GEMMs, f32 atomic segment-sum.
// A-tile staging uses GLOBAL_LOAD_ASYNC_TO_LDS_B128 (ASYNCcnt) when possible.
// ---------------------------------------------------------------------------

typedef _Float16 f16;
typedef __attribute__((ext_vector_type(16))) _Float16 v16h;
typedef __attribute__((ext_vector_type(8)))  float    v8f;

#define N_TX   100000
#define N_WL   100000
#define NEDGE  500000
#define HDIM   256

union F16Frag { uint4 u[2]; v16h v; };

// --- fp32 (K x N, row major) -> f16 transposed (N x K) weight conversion ----
__global__ void k_wconv(const float* __restrict__ W, f16* __restrict__ Wt,
                        int K, int N) {
  int i = blockIdx.x * 256 + threadIdx.x;
  if (i >= K * N) return;
  int k = i / N, n = i - k * N;
  Wt[(size_t)n * K + k] = (f16)W[i];
}

// --- per-destination edge counts -> 1/max(cnt,1) ---------------------------
__global__ void k_count(const int* __restrict__ dst, float* __restrict__ cnt, int E) {
  int i = blockIdx.x * 256 + threadIdx.x;
  if (i < E) atomicAdd(&cnt[dst[i]], 1.0f);
}
__global__ void k_invert(float* __restrict__ c, int M) {
  int i = blockIdx.x * 256 + threadIdx.x;
  if (i < M) c[i] = 1.0f / fmaxf(c[i], 1.0f);
}

// --- segment sum: one wave per edge, 256 feats = 8 f32 atomics per lane ----
__global__ void k_scatter(const f16* __restrict__ xsrc, const int* __restrict__ ei,
                          float* __restrict__ agg, int E) {
  int gt   = blockIdx.x * 256 + threadIdx.x;
  int e    = gt >> 5;
  int lane = gt & 31;
  if (e >= E) return;
  int s = ei[e];          // ei[0][e]  (source)
  int d = ei[E + e];      // ei[1][e]  (destination)
  const f16* sp = xsrc + (size_t)s * HDIM;
  float*     dp = agg  + (size_t)d * HDIM;
#pragma unroll
  for (int i = 0; i < 8; ++i) {
    int c = lane + 32 * i;
    atomicAdd(&dp[c], (float)sp[c]);
  }
}

// --- async copy of a contiguous f16 block into LDS (EXEC must be full) -----
// elems must be a multiple of 8*256 so every lane issues the same trip count.
__device__ __forceinline__ void async_stage_f16(const f16* __restrict__ src,
                                                f16* lds, int elems, int tid) {
  unsigned lbase = (unsigned)(uintptr_t)lds;   // addr[31:0] == LDS byte offset
  const int chunks = elems >> 3;               // 8 f16 = 16B per transfer
  for (int i = tid; i < chunks; i += 256) {
    unsigned la = lbase + (unsigned)(i << 4);
    unsigned long long ga = (unsigned long long)(uintptr_t)(src + (i << 3));
    asm volatile("global_load_async_to_lds_b128 %0, %1, off"
                 :: "v"(la), "v"(ga) : "memory");
  }
  asm volatile("s_wait_asynccnt 0x0" ::: "memory");
}

// --- fused WMMA GEMM: out = relu( beta*out + alpha*(A1@Wt1 + A2@Wt2 + bias) )
// A1: f32 (optionally scaled by per-row inv => "mean") or f16; A2: f16 or null.
// 16 rows per block, 8 waves, NT 16x16 column tiles per wave (N = NT*128).
template <int NT>
__global__ __launch_bounds__(256) void k_gemm(
    const void* __restrict__ A1, int a1f32, const float* __restrict__ inv1,
    const f16* __restrict__ Wt1, int K1,
    const f16* __restrict__ A2, const f16* __restrict__ Wt2, int K2,
    const float* __restrict__ bias, float alpha, int beta, int relu,
    f16* __restrict__ out, int M) {
  constexpr int NOUT = NT * 128;
  __shared__ f16 ldsA[16 * 256];   // 8 KB staging tile (16 rows x K<=256)

  const int tid  = threadIdx.x;
  const int wave = tid >> 5;
  const int lane = tid & 31;
  const int l15  = lane & 15;
  const int ks   = lane >> 4;      // K-half select per WMMA 16-bit layout
  const int row0 = blockIdx.x * 16;
  if (row0 >= M) return;

  v8f acc[NT] = {};

  for (int phase = 0; phase < 2; ++phase) {
    const int K = phase ? K2 : K1;
    if (K == 0) continue;
    const f16* Wt = phase ? Wt2 : Wt1;

    __syncthreads();
    if (phase == 0 && a1f32) {
      // f32 source needs per-row scale + convert: VALU staging path.
      const float* Af = (const float*)A1;
      for (int i = tid; i < 16 * K; i += 256) {
        int r = i / K, k = i - r * K;
        float s = inv1 ? inv1[row0 + r] : 1.0f;   // sum -> mean
        ldsA[i] = (f16)(Af[(size_t)(row0 + r) * K + k] * s);
      }
    } else {
      const f16* Ah = phase ? A2 : (const f16*)A1;
      const f16* src = Ah + (size_t)row0 * K;     // 16xK tile is contiguous
      if ((K & 127) == 0) {
        async_stage_f16(src, ldsA, 16 * K, tid);  // ASYNCcnt DMA path
      } else {
        for (int i = tid; i < 16 * K; i += 256) ldsA[i] = src[i];
      }
    }
    __syncthreads();

    for (int k0 = 0; k0 < K; k0 += 32) {
      // A fragment: row = l15, elems 0..7 -> K = k0+ks*8 .. +7,
      //             elems 8..15 -> K = k0+16+ks*8 .. +7  (contiguous 16B each)
      F16Frag a;
      const int ab = l15 * K + k0 + ks * 8;
      a.u[0] = *(const uint4*)&ldsA[ab];
      a.u[1] = *(const uint4*)&ldsA[ab + 16];
#pragma unroll
      for (int t = 0; t < NT; ++t) {
        const int n = (wave * NT + t) * 16 + l15;
        const f16* bp = Wt + (size_t)n * K + k0 + ks * 8;
        F16Frag b;
        b.u[0] = *(const uint4*)bp;
        b.u[1] = *(const uint4*)(bp + 16);
        acc[t] = __builtin_amdgcn_wmma_f32_16x16x32_f16(
            false, a.v, false, b.v, (short)0, acc[t], false, false);
      }
    }
  }

  // Epilogue: C VGPR i holds (row0 + ks*8 + i, col)
#pragma unroll
  for (int t = 0; t < NT; ++t) {
    const int col = (wave * NT + t) * 16 + l15;
    const float bia = bias ? bias[col] : 0.0f;
#pragma unroll
    for (int i = 0; i < 8; ++i) {
      const int row = row0 + ks * 8 + i;
      float v = alpha * (acc[t][i] + bia);
      const size_t o = (size_t)row * NOUT + col;
      if (beta) v += (float)out[o];
      if (relu) v = fmaxf(v, 0.0f);
      out[o] = (f16)v;
    }
  }
}

// --- tiny 128->2 head: one thread per row -----------------------------------
__global__ void k_logits(const f16* __restrict__ h, const float* __restrict__ W2,
                         const float* __restrict__ b2, float* __restrict__ out, int M) {
  int r = blockIdx.x * 256 + threadIdx.x;
  if (r >= M) return;
  float o0 = b2[0], o1 = b2[1];
  const f16* hp = h + (size_t)r * 128;
#pragma unroll 4
  for (int k = 0; k < 128; ++k) {
    float v = (float)hp[k];
    o0 += v * W2[2 * k];
    o1 += v * W2[2 * k + 1];
  }
  out[2 * r]     = o0;
  out[2 * r + 1] = o1;
}

// ---------------------------------------------------------------------------
extern "C" void kernel_launch(void* const* d_in, const int* in_sizes, int n_in,
                              void* d_out, int out_size, void* d_ws, size_t ws_size,
                              hipStream_t stream) {
  (void)in_sizes; (void)n_in; (void)out_size; (void)ws_size;

  // Top-level inputs in setup_inputs() dict order.
  const float* x_tx  = (const float*)d_in[0];
  const float* x_wl  = (const float*)d_in[1];
  const int*   ei_tt = (const int*)d_in[2];
  const int*   ei_ww = (const int*)d_in[3];
  const int*   ei_wt = (const int*)d_in[4];
  const int*   ei_tw = (const int*)d_in[5];

  // params as jax-pytree leaves (dict keys sorted):
  // layers(list of 2)[rel sorted: tt,tw,wt,ww]{Wl,Wr,bl}, tx_cls{W1,W2,b1,b2},
  // tx_lin_b, tx_lin_w, w_cls{W1,W2,b1,b2}, w_lin_b, w_lin_w
  const float *Wl[2][4], *Wr[2][4], *bl[2][4];
  int p = 6;
  for (int l = 0; l < 2; ++l)
    for (int r = 0; r < 4; ++r) {
      Wl[l][r] = (const float*)d_in[p++];
      Wr[l][r] = (const float*)d_in[p++];
      bl[l][r] = (const float*)d_in[p++];
    }
  const float* txc_W1 = (const float*)d_in[p++];
  const float* txc_W2 = (const float*)d_in[p++];
  const float* txc_b1 = (const float*)d_in[p++];
  const float* txc_b2 = (const float*)d_in[p++];
  const float* tx_lin_b = (const float*)d_in[p++];
  const float* tx_lin_w = (const float*)d_in[p++];
  const float* wc_W1 = (const float*)d_in[p++];
  const float* wc_W2 = (const float*)d_in[p++];
  const float* wc_b1 = (const float*)d_in[p++];
  const float* wc_b2 = (const float*)d_in[p++];
  const float* w_lin_b = (const float*)d_in[p++];
  const float* w_lin_w = (const float*)d_in[p++];

  // Workspace carve (256B aligned suballocations).
  char* wp = (char*)d_ws;
  auto alloc = [&](size_t bytes) -> char* {
    char* r = wp;
    wp += (bytes + 255) & ~(size_t)255;
    return r;
  };
  f16* txA = (f16*)alloc((size_t)N_TX * HDIM * 2);
  f16* txB = (f16*)alloc((size_t)N_TX * HDIM * 2);
  f16* wlA = (f16*)alloc((size_t)N_WL * HDIM * 2);
  f16* wlB = (f16*)alloc((size_t)N_WL * HDIM * 2);
  float* agg = (float*)alloc((size_t)N_TX * HDIM * 4);   // f32 segment-sum buffer
  float* invb[4];
  for (int r = 0; r < 4; ++r) invb[r] = (float*)alloc((size_t)N_TX * 4);
  f16 *Wlt[2][4], *Wrt[2][4];
  for (int l = 0; l < 2; ++l)
    for (int r = 0; r < 4; ++r) {
      Wlt[l][r] = (f16*)alloc((size_t)HDIM * HDIM * 2);
      Wrt[l][r] = (f16*)alloc((size_t)HDIM * HDIM * 2);
    }
  f16* txlin_t = (f16*)alloc((size_t)256 * 64 * 2);
  f16* wlin_t  = (f16*)alloc((size_t)256 * 32 * 2);
  f16* txcW1t  = (f16*)alloc((size_t)128 * 256 * 2);
  f16* wcW1t   = (f16*)alloc((size_t)128 * 256 * 2);
  f16* hbuf    = (f16*)agg;   // classifier hidden reuses agg (free by then)

  auto conv = [&](const float* src, f16* dst, int K, int N) {
    int tot = K * N;
    k_wconv<<<(tot + 255) / 256, 256, 0, stream>>>(src, dst, K, N);
  };
  for (int l = 0; l < 2; ++l)
    for (int r = 0; r < 4; ++r) {
      conv(Wl[l][r], Wlt[l][r], 256, 256);
      conv(Wr[l][r], Wrt[l][r], 256, 256);
    }
  conv(tx_lin_w, txlin_t, 64, 256);
  conv(w_lin_w, wlin_t, 32, 256);
  conv(txc_W1, txcW1t, 256, 128);
  conv(wc_W1, wcW1t, 256, 128);

  // Edge counts -> reciprocals (invariant across layers). rel order tt,tw,wt,ww.
  const int* eis[4] = {ei_tt, ei_tw, ei_wt, ei_ww};
  for (int r = 0; r < 4; ++r) {
    hipMemsetAsync(invb[r], 0, (size_t)N_TX * 4, stream);
    k_count<<<(NEDGE + 255) / 256, 256, 0, stream>>>(eis[r] + NEDGE, invb[r], NEDGE);
    k_invert<<<(N_TX + 255) / 256, 256, 0, stream>>>(invb[r], N_TX);
  }

  // Input projections (no relu).
  k_gemm<2><<<N_TX / 16, 256, 0, stream>>>(x_tx, 1, nullptr, txlin_t, 64,
      nullptr, nullptr, 0, tx_lin_b, 1.0f, 0, 0, txA, N_TX);
  k_gemm<2><<<N_WL / 16, 256, 0, stream>>>(x_wl, 1, nullptr, wlin_t, 32,
      nullptr, nullptr, 0, w_lin_b, 1.0f, 0, 0, wlA, N_WL);

  const size_t aggBytes = (size_t)N_TX * HDIM * 4;
  const int scatterGrid = (NEDGE * 32 + 255) / 256;
  f16 *ctx = txA, *ntx = txB, *cwl = wlA, *nwl = wlB;

  for (int l = 0; l < 2; ++l) {
    const int relu = (l == 0);
    // dst = tx : relation tt (src tx), then wt (src wallet)
    hipMemsetAsync(agg, 0, aggBytes, stream);
    k_scatter<<<scatterGrid, 256, 0, stream>>>(ctx, ei_tt, agg, NEDGE);
    k_gemm<2><<<N_TX / 16, 256, 0, stream>>>(agg, 1, invb[0], Wlt[l][0], 256,
        ctx, Wrt[l][0], 256, bl[l][0], 0.5f, 0, 0, ntx, N_TX);
    hipMemsetAsync(agg, 0, aggBytes, stream);
    k_scatter<<<scatterGrid, 256, 0, stream>>>(cwl, ei_wt, agg, NEDGE);
    k_gemm<2><<<N_TX / 16, 256, 0, stream>>>(agg, 1, invb[2], Wlt[l][2], 256,
        ctx, Wrt[l][2], 256, bl[l][2], 0.5f, 1, relu, ntx, N_TX);
    // dst = wallet : relation ww (src wallet), then tw (src tx; ctx still intact)
    hipMemsetAsync(agg, 0, aggBytes, stream);
    k_scatter<<<scatterGrid, 256, 0, stream>>>(cwl, ei_ww, agg, NEDGE);
    k_gemm<2><<<N_WL / 16, 256, 0, stream>>>(agg, 1, invb[3], Wlt[l][3], 256,
        cwl, Wrt[l][3], 256, bl[l][3], 0.5f, 0, 0, nwl, N_WL);
    hipMemsetAsync(agg, 0, aggBytes, stream);
    k_scatter<<<scatterGrid, 256, 0, stream>>>(ctx, ei_tw, agg, NEDGE);
    k_gemm<2><<<N_WL / 16, 256, 0, stream>>>(agg, 1, invb[1], Wlt[l][1], 256,
        cwl, Wrt[l][1], 256, bl[l][1], 0.5f, 1, relu, nwl, N_WL);
    f16* t;
    t = ctx; ctx = ntx; ntx = t;
    t = cwl; cwl = nwl; nwl = t;
  }

  // Classifier heads: 256 -> 128 (relu, WMMA) -> 2 (scalar).
  float* out_tx = (float*)d_out;
  float* out_wl = (float*)d_out + (size_t)2 * N_TX;
  k_gemm<1><<<N_TX / 16, 256, 0, stream>>>(ctx, 0, nullptr, txcW1t, 256,
      nullptr, nullptr, 0, txc_b1, 1.0f, 0, 1, hbuf, N_TX);
  k_logits<<<(N_TX + 255) / 256, 256, 0, stream>>>(hbuf, txc_W2, txc_b2, out_tx, N_TX);
  k_gemm<1><<<N_WL / 16, 256, 0, stream>>>(cwl, 0, nullptr, wcW1t, 256,
      nullptr, nullptr, 0, wc_b1, 1.0f, 0, 1, hbuf, N_WL);
  k_logits<<<(N_WL + 255) / 256, 256, 0, stream>>>(hbuf, wc_W2, wc_b2, out_wl, N_WL);
}
